// RGCNAggregator_global_33526514713102
// MI455X (gfx1250) — matrix operand order repaired and checked by reading, not compiled, not run on hardware
//
#include <hip/hip_runtime.h>
#include <hip/hip_bf16.h>

// ---------------- problem constants ----------------
#define NUM_ENTS 20000
#define H        200
#define HP       224          // H padded to multiple of 32 (WMMA K-chunk)
#define NUM_REL2 460
#define N_NODES  252000
#define N_EDGES  1008000
#define T_GRAPH  63
#define B_BATCH  64
#define S_SEQ    10
#define N_TILES  (N_NODES / 16)     // 15750, exact
#define TILES_PER_BLOCK 16          // 512 threads = 16 waves
#define W_LDS_DWORDS (2 * HP * HP / 2)   // 50176 dwords = 200704 B (2 bf16 matrices)

typedef __attribute__((ext_vector_type(16))) __bf16 v16bf;
typedef __attribute__((ext_vector_type(8)))  float  v8f;
typedef __attribute__((ext_vector_type(4)))  unsigned int u32x4;
typedef __attribute__((ext_vector_type(8)))  int i32x8;
typedef __attribute__((ext_vector_type(4)))  int i32x4;

// ---------------- TDM: copy the whole 200,704 B weight block (50176 dwords) to LDS.
// D# per CDNA5 ISA ch.8: group0 = {count=1 | lds_addr | global_addr | type=2},
// group1 = {data_size=4B, tensor_dim0=tile_dim0=50176, tensor_dim1=tile_dim1=1,
//           dim0_stride=50176}. Groups 2/3 zero (2D tensor).
__device__ __forceinline__ void tdm_load_w_to_lds(const __bf16* Wt) {
    unsigned long long ga = (unsigned long long)(uintptr_t)Wt;
    u32x4 g0 = { 1u,                                   // count=1, user mode, no gather
                 0u,                                   // lds_addr = 0 (start of dyn LDS)
                 (unsigned)(ga & 0xFFFFFFFFull),       // global_addr[31:0]
                 (unsigned)((ga >> 32) & 0x01FFFFFFull) | (2u << 30) };  // [56:32] | type=2
    i32x8 g1 = { (int)0x20000,                         // workgroup_mask=0, data_size=2 (4B)
                 (int)((unsigned)W_LDS_DWORDS << 16),  // tensor_dim0[15:0] (=50176)
                 (int)(1u << 16),                      // tensor_dim0[31:16]=0, tensor_dim1=1
                 (int)((unsigned)W_LDS_DWORDS << 16),  // tensor_dim1 hi=0, tile_dim0=50176
                 1,                                    // tile_dim1=1, tile_dim2=0
                 W_LDS_DWORDS,                         // tensor_dim0_stride[31:0]
                 0, 0 };                               // stride hi / dim1_stride = 0
    i32x4 gz = { 0, 0, 0, 0 };
#if defined(__clang_major__) && (__clang_major__ >= 23)
    i32x8 gz8 = { 0, 0, 0, 0, 0, 0, 0, 0 };
    __builtin_amdgcn_tensor_load_to_lds(g0, g1, gz, gz, gz8, 0);
#else
    __builtin_amdgcn_tensor_load_to_lds(g0, g1, gz, gz, 0);
#endif
}

// ---------------- prep: Wt[m][n][k] = W_m[k*H + n] (bf16, zero-padded to 224x224)
__global__ __launch_bounds__(256) void prep_w_kernel(const float* __restrict__ Wmsg,
                                                     const float* __restrict__ Wloop,
                                                     __bf16* __restrict__ Wt) {
    int gid = blockIdx.x * 256 + threadIdx.x;      // 2*224*224 = 100352
    if (gid >= 2 * HP * HP) return;
    int m = gid / (HP * HP);
    int r = gid - m * (HP * HP);
    int n = r / HP;
    int k = r - n * HP;
    float v = 0.0f;
    if (n < H && k < H) {
        const float* W = (m == 0) ? Wmsg : Wloop;
        v = W[k * H + n];
    }
    Wt[gid] = (__bf16)v;
}

// ---------------- node init: h[n, c] = ent_embeds[node_ids[n], c], pad cols zero
__global__ __launch_bounds__(HP) void init_h_kernel(const float* __restrict__ ent,
                                                    const int* __restrict__ node_ids,
                                                    float* __restrict__ h) {
    int n = blockIdx.x;            // one block per node, 224 threads
    int c = threadIdx.x;
    float v = 0.0f;
    if (c < H) v = ent[(size_t)node_ids[n] * H + c];
    h[(size_t)n * HP + c] = v;
}

// ---------------- zero a float buffer (count divisible by 4)
__global__ __launch_bounds__(256) void zero_kernel(float4* __restrict__ p, int n4) {
    int gid = blockIdx.x * 256 + threadIdx.x;
    if (gid < n4) p[gid] = make_float4(0.f, 0.f, 0.f, 0.f);
}

// ---------------- edge scatter: tmp[dst] += h[src] + rel_embeds[rel]
__global__ __launch_bounds__(256) void scatter_kernel(const float* __restrict__ h,
                                                      const float* __restrict__ rel,
                                                      const int* __restrict__ esrc,
                                                      const int* __restrict__ edst,
                                                      const int* __restrict__ erel,
                                                      float* __restrict__ tmp) {
    int gid = blockIdx.x * 256 + threadIdx.x;      // E * 50 work items
    int edge = gid / 50;
    int chunk = gid - edge * 50;                   // float4 chunk, covers cols 0..199
    if (edge >= N_EDGES) return;
    int s = esrc[edge];
    int d = edst[edge];
    int r = erel[edge];
    const float4 hv = *(const float4*)(h + (size_t)s * HP + chunk * 4);
    const float4 rv = *(const float4*)(rel + (size_t)r * H + chunk * 4);
    float* o = tmp + (size_t)d * HP + chunk * 4;
    atomicAdd(o + 0, hv.x + rv.x);
    atomicAdd(o + 1, hv.y + rv.y);
    atomicAdd(o + 2, hv.z + rv.z);
    atomicAdd(o + 3, hv.w + rv.w);
}

// ---------------- fused GEMM: tmp = relu(tmp @ Wmsg + h @ Wloop), in place on tmp.
// 512 threads = 16 waves, one 16-row stripe per wave. Weights staged in LDS by the
// Tensor Data Mover (one 200KB descriptor), shared by all 16 waves; A fragments are
// register-resident (f32 -> bf16 cvt). 196 v_wmma_f32_16x16x32_bf16 per wave, B
// fragments from ds_load_b128.
__global__ __launch_bounds__(512) void gemm_relu_kernel(float* __restrict__ tmpbuf,
                                                        const float* __restrict__ hbuf,
                                                        const __bf16* __restrict__ Wt) {
    extern __shared__ __bf16 ldsW[];               // 2*224*224 bf16 = 200,704 B

    const int wave = threadIdx.x >> 5;
    const int lane = threadIdx.x & 31;
    const int tile = blockIdx.x * TILES_PER_BLOCK + wave;

    if (wave == 0) tdm_load_w_to_lds(Wt);          // async: overlaps the A loads below

    const int row0 = tile * 16;
    const int arow = row0 + (lane & 15);           // A: M = lane&15
    const int khi  = (lane >> 4) * 16;             // A: lanes>=16 hold K+16 block

    v16bf a0[7], a1[7];
    if (tile < N_TILES) {
        #pragma unroll
        for (int kc = 0; kc < 7; ++kc) {
            const int kb = kc * 32 + khi;
            const float* p0 = tmpbuf + (size_t)arow * HP + kb;
            const float* p1 = hbuf   + (size_t)arow * HP + kb;
            float4 q0 = *(const float4*)(p0);
            float4 q1 = *(const float4*)(p0 + 4);
            float4 q2 = *(const float4*)(p0 + 8);
            float4 q3 = *(const float4*)(p0 + 12);
            v16bf a;
            a[0]  = (__bf16)q0.x; a[1]  = (__bf16)q0.y; a[2]  = (__bf16)q0.z; a[3]  = (__bf16)q0.w;
            a[4]  = (__bf16)q1.x; a[5]  = (__bf16)q1.y; a[6]  = (__bf16)q1.z; a[7]  = (__bf16)q1.w;
            a[8]  = (__bf16)q2.x; a[9]  = (__bf16)q2.y; a[10] = (__bf16)q2.z; a[11] = (__bf16)q2.w;
            a[12] = (__bf16)q3.x; a[13] = (__bf16)q3.y; a[14] = (__bf16)q3.z; a[15] = (__bf16)q3.w;
            a0[kc] = a;
            q0 = *(const float4*)(p1);
            q1 = *(const float4*)(p1 + 4);
            q2 = *(const float4*)(p1 + 8);
            q3 = *(const float4*)(p1 + 12);
            a[0]  = (__bf16)q0.x; a[1]  = (__bf16)q0.y; a[2]  = (__bf16)q0.z; a[3]  = (__bf16)q0.w;
            a[4]  = (__bf16)q1.x; a[5]  = (__bf16)q1.y; a[6]  = (__bf16)q1.z; a[7]  = (__bf16)q1.w;
            a[8]  = (__bf16)q2.x; a[9]  = (__bf16)q2.y; a[10] = (__bf16)q2.z; a[11] = (__bf16)q2.w;
            a[12] = (__bf16)q3.x; a[13] = (__bf16)q3.y; a[14] = (__bf16)q3.z; a[15] = (__bf16)q3.w;
            a1[kc] = a;
        }
    }

    if (wave == 0) __builtin_amdgcn_s_wait_tensorcnt(0);   // TDM data landed in LDS
    __syncthreads();                                       // make it visible to all waves
    if (tile >= N_TILES) return;                           // wave-uniform

    const int ncol  = lane & 15;                   // B/D: N = lane&15
    const int kwsel = (lane >> 4) * 16;            // B: lanes>=16 hold K+16 block
    const __bf16* W0 = ldsW;                       // Wmsg^T, [224][224] in LDS
    const __bf16* W1 = ldsW + HP * HP;             // Wloop^T in LDS
    const int mbase = row0 + ((lane >> 4) << 3);   // D: M = (lane>=16)*8 + r

    for (int ct = 0; ct < 14; ++ct) {
        v8f acc = {};
        const int n = ct * 16 + ncol;
        #pragma unroll
        for (int kc = 0; kc < 7; ++kc) {
            v16bf b = *(const v16bf*)(W0 + n * HP + kc * 32 + kwsel);
            acc = __builtin_amdgcn_wmma_f32_16x16x32_bf16(
                false, a0[kc], false, b, (short)0, acc, false, false);
        }
        #pragma unroll
        for (int kc = 0; kc < 7; ++kc) {
            v16bf b = *(const v16bf*)(W1 + n * HP + kc * 32 + kwsel);
            acc = __builtin_amdgcn_wmma_f32_16x16x32_bf16(
                false, a1[kc], false, b, (short)0, acc, false, false);
        }
        const int col = ct * 16 + ncol;
        #pragma unroll
        for (int r = 0; r < 8; ++r) {
            float v = acc[r];
            v = (v > 0.0f) ? v : 0.0f;             // relu
            tmpbuf[(size_t)(mbase + r) * HP + col] = v;
        }
    }
}

// ---------------- per-graph max pool: gmax[t] = max over nodes (values >= 0)
__global__ __launch_bounds__(256) void segmax_kernel(const float* __restrict__ h,
                                                     const int* __restrict__ n2g,
                                                     float* __restrict__ gmax) {
    int gid = blockIdx.x * 256 + threadIdx.x;      // N * 50 work items
    int n = gid / 50;
    int chunk = gid - n * 50;
    if (n >= N_NODES) return;
    int t = n2g[n];
    const float4 v = *(const float4*)(h + (size_t)n * HP + chunk * 4);
    unsigned int* g = (unsigned int*)(gmax + (size_t)t * HP + chunk * 4);
    atomicMax(g + 0, __float_as_uint(v.x));        // valid: all values >= 0
    atomicMax(g + 1, __float_as_uint(v.y));
    atomicMax(g + 2, __float_as_uint(v.z));
    atomicMax(g + 3, __float_as_uint(v.w));
}

// ---------------- final padded gather: out[b,s,:] = mask ? gmax[time_idx[b,s]] : 0
__global__ __launch_bounds__(256) void gather_out_kernel(const float* __restrict__ gmax,
                                                         const int* __restrict__ time_idx,
                                                         const float* __restrict__ seq_mask,
                                                         float* __restrict__ out) {
    int gid = blockIdx.x * 256 + threadIdx.x;      // B*S*H = 128000
    if (gid >= B_BATCH * S_SEQ * H) return;
    int bs = gid / H;
    int c  = gid - bs * H;
    float v = 0.0f;
    if (seq_mask[bs] > 0.0f) v = gmax[(size_t)time_idx[bs] * HP + c];
    out[gid] = v;
}

// ---------------- launcher ----------------
extern "C" void kernel_launch(void* const* d_in, const int* in_sizes, int n_in,
                              void* d_out, int out_size, void* d_ws, size_t ws_size,
                              hipStream_t stream) {
    const float* ent_embeds = (const float*)d_in[0];
    const float* rel_embeds = (const float*)d_in[1];
    const float* W_msg      = (const float*)d_in[2];
    const float* W_loop     = (const float*)d_in[3];
    const int*   node_ids   = (const int*)d_in[4];
    const int*   edge_src   = (const int*)d_in[5];
    const int*   edge_dst   = (const int*)d_in[6];
    const int*   edge_rel   = (const int*)d_in[7];
    const int*   node2graph = (const int*)d_in[8];
    const int*   time_idx   = (const int*)d_in[9];
    const float* seq_mask   = (const float*)d_in[10];
    float* out = (float*)d_out;

    // workspace layout (all offsets 256B aligned)
    const size_t nodeBufBytes = (size_t)N_NODES * HP * sizeof(float);   // 225,792,000
    char* ws = (char*)d_ws;
    float*  buf0 = (float*)(ws);
    float*  buf1 = (float*)(ws + nodeBufBytes);
    __bf16* Wt   = (__bf16*)(ws + 2 * nodeBufBytes);                    // 2*224*224 bf16
    float*  gmax = (float*)(ws + 2 * nodeBufBytes + (size_t)2 * HP * HP * sizeof(__bf16));

    const int nodeBufFloats = N_NODES * HP;               // 56,448,000
    const int gmaxFloats    = 64 * HP;                    // >= T_GRAPH*HP, /4 ok
    const size_t wLdsBytes  = (size_t)2 * HP * HP * sizeof(__bf16);     // 200,704

    // prep
    prep_w_kernel<<<(2 * HP * HP + 255) / 256, 256, 0, stream>>>(W_msg, W_loop, Wt);
    init_h_kernel<<<N_NODES, HP, 0, stream>>>(ent_embeds, node_ids, buf0);

    const int scatterBlocks = (N_EDGES * 50 + 255) / 256;
    const int gemmBlocks    = (N_TILES + TILES_PER_BLOCK - 1) / TILES_PER_BLOCK;
    const int zeroBlocksN   = (nodeBufFloats / 4 + 255) / 256;

    // ---- layer 1: h = buf0, tmp = buf1 ; result lands in buf1
    zero_kernel<<<zeroBlocksN, 256, 0, stream>>>((float4*)buf1, nodeBufFloats / 4);
    scatter_kernel<<<scatterBlocks, 256, 0, stream>>>(buf0, rel_embeds,
                                                      edge_src, edge_dst, edge_rel, buf1);
    gemm_relu_kernel<<<gemmBlocks, 512, wLdsBytes, stream>>>(buf1, buf0, Wt);

    // ---- layer 2: h = buf1, tmp = buf0 ; result lands in buf0
    zero_kernel<<<zeroBlocksN, 256, 0, stream>>>((float4*)buf0, nodeBufFloats / 4);
    scatter_kernel<<<scatterBlocks, 256, 0, stream>>>(buf1, rel_embeds,
                                                      edge_src, edge_dst, edge_rel, buf0);
    gemm_relu_kernel<<<gemmBlocks, 512, wLdsBytes, stream>>>(buf0, buf1, Wt);

    // ---- max pool + final gather
    zero_kernel<<<(gmaxFloats / 4 + 255) / 256, 256, 0, stream>>>((float4*)gmax, gmaxFloats / 4);
    segmax_kernel<<<(N_NODES * 50 + 255) / 256, 256, 0, stream>>>(buf0, node2graph, gmax);
    gather_out_kernel<<<(B_BATCH * S_SEQ * H + 255) / 256, 256, 0, stream>>>(gmax, time_idx,
                                                                             seq_mask, out);
}
